// VanillaRNN_43928925503590
// MI455X (gfx1250) — compile-verified, matrix operand order checked by reading, not run on hardware
//
#include <hip/hip_runtime.h>

// ---------------------------------------------------------------------------
// VanillaRNN on MI455X (gfx1250): bf16 WMMA, f32 accumulate.
//   pre-pass: x, W_hx, W_hh -> bf16 (bandwidth-bound, ~12us at 23.3 TB/s)
//   xproj = x @ W_hx^T + b_h          (parallel GEMM kernel, pure bf16 loads,
//                                      no per-wmma cvt VALU -> WMMA-bound)
//   h_t   = xproj_t + h_{t-1}@W_hh^T  (ONE persistent kernel, 128 steps,
//                                      grid-wide software barrier;
//                                      W_hh panel cached in 256KB LDS via
//                                      global_load_async_to_lds_b128)
//   p     = h_T @ W_ph^T + b_p        (folded into persistent kernel tail)
// ---------------------------------------------------------------------------

typedef __bf16 bf16_t;
typedef __attribute__((ext_vector_type(16))) __bf16 v16bf;
typedef __attribute__((ext_vector_type(8)))  __bf16 v8bf;
typedef __attribute__((ext_vector_type(8)))  float  v8f;
typedef __attribute__((ext_vector_type(4)))  float  f32x4;
typedef unsigned int u32x4 __attribute__((ext_vector_type(4)));

union FragU { v16bf v; u32x4 q[2]; };

// A/B fragment for V_WMMA_F32_16X16X32_BF16 (wave32):
//  lanes 0-15  : row m0+lane,    elems 0-7 = K[k0..+7],    8-15 = K[k0+16..+23]
//  lanes 16-31 : row m0+lane-16, elems 0-7 = K[k0+8..+15], 8-15 = K[k0+24..+31]
__device__ __forceinline__ v16bf load_frag(const bf16_t* __restrict__ row, int k0, int half) {
  FragU u;
  u.q[0] = *(const u32x4*)(row + k0 + 8 * half);
  u.q[1] = *(const u32x4*)(row + k0 + 16 + 8 * half);
  return u.v;
}

// f32 source -> convert in-register; only used for W_ph in the tiny readout.
__device__ __forceinline__ v16bf load_frag(const float* __restrict__ row, int k0, int half) {
  const float* p0 = row + k0 + 8 * half;
  const float* p1 = row + k0 + 16 + 8 * half;
  v16bf v;
#pragma unroll
  for (int i = 0; i < 8; ++i) {
    v[i]     = (__bf16)p0[i];
    v[i + 8] = (__bf16)p1[i];
  }
  return v;
}

// ---------------- parallel GEMM (used for xproj): C = A*B^T (+bias) ---------
template <typename AT, typename BT, typename OT, bool HAS_BIAS,
          int WM_WAVES, int WN_WAVES, int WN_TILES>
__global__ __launch_bounds__(32 * WM_WAVES * WN_WAVES)
void wmma_gemm_abT(const AT* __restrict__ A, int lda,
                   const BT* __restrict__ B, int ldb,
                   OT* __restrict__ C, int ldc,
                   const float* __restrict__ bias, int K) {
  const int tid  = threadIdx.x;
  const int wave = tid >> 5;
  const int lane = tid & 31;
  const int r    = lane & 15;
  const int half = lane >> 4;
  const int wm   = wave % WM_WAVES;
  const int wn   = wave / WM_WAVES;
  const int m0   = blockIdx.y * (WM_WAVES * 16) + wm * 16;
  const int n0   = blockIdx.x * (WN_WAVES * WN_TILES * 16) + wn * (WN_TILES * 16);

  const AT* arow = A + (size_t)(m0 + r) * lda;
  const BT* brow[WN_TILES];
#pragma unroll
  for (int t = 0; t < WN_TILES; ++t)
    brow[t] = B + (size_t)(n0 + t * 16 + r) * ldb;

  v8f acc[WN_TILES];
#pragma unroll
  for (int t = 0; t < WN_TILES; ++t)
#pragma unroll
    for (int j = 0; j < 8; ++j) acc[t][j] = 0.0f;

  for (int k0 = 0; k0 < K; k0 += 32) {
    v16bf af = load_frag(arow, k0, half);
#pragma unroll
    for (int t = 0; t < WN_TILES; ++t) {
      v16bf bfr = load_frag(brow[t], k0, half);
      acc[t] = __builtin_amdgcn_wmma_f32_16x16x32_bf16(
          false, af, false, bfr, (short)0, acc[t], false, false);
    }
  }

#pragma unroll
  for (int t = 0; t < WN_TILES; ++t)
#pragma unroll
    for (int j = 0; j < 8; ++j) {
      const int cm = m0 + j + 8 * half;
      const int cn = n0 + t * 16 + r;
      float v = acc[t][j];
      if (HAS_BIAS) v += bias[cn];
      C[(size_t)cm * ldc + cn] = (OT)v;
    }
}

// Streaming f32 -> bf16: 8 elements / thread / iter (32B in, 16B out).
__global__ void cvt_f32_to_bf16_x8(const float* __restrict__ in,
                                   bf16_t* __restrict__ out, int n8) {
  for (int i = blockIdx.x * blockDim.x + threadIdx.x; i < n8;
       i += gridDim.x * blockDim.x) {
    const f32x4* p = (const f32x4*)(in + (size_t)i * 8);
    f32x4 a = p[0], b = p[1];
    v8bf o;
#pragma unroll
    for (int j = 0; j < 4; ++j) {
      o[j]     = (__bf16)a[j];
      o[j + 4] = (__bf16)b[j];
    }
    *(v8bf*)(out + (size_t)i * 8) = o;
  }
}

// ---------------- persistent recurrence + readout ---------------------------
__device__ __forceinline__ void grid_barrier(unsigned* bar, unsigned target) {
  __syncthreads();
  if (threadIdx.x == 0) {
    __threadfence();                   // release h stores to device scope
    atomicAdd(bar, 1u);
    while (atomicAdd(bar, 0u) < target)
      __builtin_amdgcn_s_sleep(2);
    __threadfence();                   // acquire other blocks' h stores
  }
  __syncthreads();
}

// grid = 128 blocks x 256 threads; each block owns a 64(M) x 64(N) tile of h.
// Requires all 128 blocks co-resident (1 block/WGP at 256KB LDS; MI455X-class
// parts have >=128 WGPs).
__global__ __launch_bounds__(256)
void rnn_recurrence_readout(const bf16_t* __restrict__ Whh,    // [2048][2048] bf16
                            const bf16_t* __restrict__ xproj,  // [B][S][H] bf16
                            const float*  __restrict__ Wph,    // [1024][2048] f32
                            const float*  __restrict__ b_p,
                            bf16_t* __restrict__ h0,           // pre-zeroed
                            bf16_t* __restrict__ h1,
                            float*  __restrict__ out,          // [B][C]
                            unsigned* __restrict__ bar) {
  constexpr int B_ = 256, S_ = 128, H_ = 2048, C_ = 1024;
  extern __shared__ char smem_raw[];
  bf16_t* bpan = (bf16_t*)smem_raw;          // [64][2048] = 256 KB panel of W_hh

  const int bid  = blockIdx.x;               // 0..127
  const int nblk = gridDim.x;
  const int tid  = threadIdx.x;
  const int wave = tid >> 5, lane = tid & 31;
  const int r = lane & 15, half = lane >> 4;
  const int wm = wave & 3, wn = wave >> 2;   // 4 x 2 waves, wave tile 16x32
  const int bx = bid & 31, by = bid >> 5;    // N blocks = 2048/64, M blocks = 256/64
  const int m0  = by * 64 + wm * 16;
  const int n0b = bx * 64;
  const int n0  = n0b + wn * 32;

  // -- one-time: async-copy this block's 64-row W_hh panel (256 KB) into LDS --
  {
    const unsigned lbase = (unsigned)(uintptr_t)(void*)bpan;
    const bf16_t* gbase  = Whh + (size_t)n0b * H_;   // contiguous 256 KB
    for (int c = tid; c < (64 * H_) / 8; c += 256) { // 16-byte chunks
      unsigned laddr   = lbase + (unsigned)c * 16u;
      const bf16_t* g  = gbase + (size_t)c * 8;
      asm volatile("global_load_async_to_lds_b128 %0, %1, off"
                   :: "v"(laddr), "v"(g) : "memory");
    }
    asm volatile("s_wait_asynccnt 0" ::: "memory");
    __syncthreads();
  }

  // -------------------- 128 recurrence steps --------------------
  for (int t = 0; t < S_; ++t) {
    const bf16_t* hin    = (t & 1) ? h1 : h0;
    bf16_t*       hout   = (t & 1) ? h0 : h1;
    const bf16_t* init_t = xproj + (size_t)t * H_;   // element (b,h) at b*S*H + h

    // init accumulators from xproj[:, t, :]
    v8f acc[2];
#pragma unroll
    for (int tt = 0; tt < 2; ++tt)
#pragma unroll
      for (int j = 0; j < 8; ++j) {
        const int cm = m0 + j + 8 * half;
        const int cn = n0 + tt * 16 + r;
        acc[tt][j] = (float)init_t[(size_t)cm * (S_ * H_) + cn];
      }

    // prefetch next step's xproj slice into cache (global_prefetch_b8)
    if (t + 1 < S_)
      __builtin_prefetch(init_t + H_ + (size_t)(m0 + r) * (size_t)(S_ * H_) + n0, 0, 1);

    const bf16_t* arow = hin + (size_t)(m0 + r) * H_;
    for (int k0 = 0; k0 < H_; k0 += 32) {
      v16bf af = load_frag(arow, k0, half);
#pragma unroll
      for (int tt = 0; tt < 2; ++tt) {
        const bf16_t* brow = bpan + (size_t)(wn * 32 + tt * 16 + r) * H_;  // LDS
        v16bf bfr = load_frag(brow, k0, half);     // ds_load_b128 x2
        acc[tt] = __builtin_amdgcn_wmma_f32_16x16x32_bf16(
            false, af, false, bfr, (short)0, acc[tt], false, false);
      }
    }

#pragma unroll
    for (int tt = 0; tt < 2; ++tt)
#pragma unroll
      for (int j = 0; j < 8; ++j) {
        const int cm = m0 + j + 8 * half;
        const int cn = n0 + tt * 16 + r;
        hout[(size_t)cm * H_ + cn] = (bf16_t)acc[tt][j];
      }

    grid_barrier(bar, (unsigned)((t + 1) * nblk));
  }

  // -------------------- readout: p = h_T @ W_ph^T + b_p --------------------
  // S_=128 (even) -> last write went to h0. 64 output tiles -> blocks 0..63.
  if (bid < (B_ / 64) * (C_ / 64)) {
    const int rbx = bid & 15, rby = bid >> 4;      // C/64 = 16
    const int rm0 = rby * 64 + wm * 16;
    const int rn0 = rbx * 64 + wn * 32;
    const bf16_t* arow = h0 + (size_t)(rm0 + r) * H_;

    v8f acc[2];
#pragma unroll
    for (int tt = 0; tt < 2; ++tt)
#pragma unroll
      for (int j = 0; j < 8; ++j) acc[tt][j] = 0.0f;

    for (int k0 = 0; k0 < H_; k0 += 32) {
      v16bf af = load_frag(arow, k0, half);
#pragma unroll
      for (int tt = 0; tt < 2; ++tt) {
        const float* brow = Wph + (size_t)(rn0 + tt * 16 + r) * H_;  // f32, cvt inline
        v16bf bfr = load_frag(brow, k0, half);
        acc[tt] = __builtin_amdgcn_wmma_f32_16x16x32_bf16(
            false, af, false, bfr, (short)0, acc[tt], false, false);
      }
    }

#pragma unroll
    for (int tt = 0; tt < 2; ++tt)
#pragma unroll
      for (int j = 0; j < 8; ++j) {
        const int cm = rm0 + j + 8 * half;
        const int cn = rn0 + tt * 16 + r;
        out[(size_t)cm * C_ + cn] = acc[tt][j] + b_p[cn];
      }
  }
}

// ---------------------------------------------------------------------------

extern "C" void kernel_launch(void* const* d_in, const int* in_sizes, int n_in,
                              void* d_out, int out_size, void* d_ws, size_t ws_size,
                              hipStream_t stream) {
  (void)in_sizes; (void)n_in; (void)out_size; (void)ws_size;
  constexpr int BATCH = 256, SEQ = 128, D_IN = 1024, D_HID = 2048, N_CLS = 1024;
  constexpr size_t LDS_BYTES = 64 * (size_t)D_HID * sizeof(bf16_t);  // 256 KB

  const float* x    = (const float*)d_in[0];
  const float* W_hx = (const float*)d_in[1];
  const float* W_hh = (const float*)d_in[2];
  const float* W_ph = (const float*)d_in[3];
  const float* b_h  = (const float*)d_in[4];
  const float* b_p  = (const float*)d_in[5];
  float* out = (float*)d_out;

  // Workspace layout (256B aligned by construction; total ~206 MiB):
  char* ws = (char*)d_ws;
  size_t off = 0;
  bf16_t* Whh_bf = (bf16_t*)(ws + off); off += (size_t)D_HID * D_HID * sizeof(bf16_t);   // 8 MiB
  bf16_t* xproj  = (bf16_t*)(ws + off); off += (size_t)BATCH * SEQ * D_HID * sizeof(bf16_t); // 128 MiB
  bf16_t* h0     = (bf16_t*)(ws + off); off += (size_t)BATCH * D_HID * sizeof(bf16_t);   // 1 MiB
  bf16_t* h1     = (bf16_t*)(ws + off); off += (size_t)BATCH * D_HID * sizeof(bf16_t);   // 1 MiB
  unsigned* bar  = (unsigned*)(ws + off); off += 256;
  bf16_t* x_bf   = (bf16_t*)(ws + off); off += (size_t)BATCH * SEQ * D_IN * sizeof(bf16_t); // 64 MiB
  bf16_t* Whx_bf = (bf16_t*)(ws + off); off += (size_t)D_HID * D_IN * sizeof(bf16_t);    // 4 MiB

  // Allow 256 KB dynamic LDS for the persistent kernel.
  hipFuncSetAttribute((const void*)&rnn_recurrence_readout,
                      hipFuncAttributeMaxDynamicSharedMemorySize, (int)LDS_BYTES);

  // 1) Streaming f32 -> bf16 pre-conversion (x, W_hx, W_hh).
  cvt_f32_to_bf16_x8<<<2048, 256, 0, stream>>>(x, x_bf, (BATCH * SEQ * D_IN) / 8);
  cvt_f32_to_bf16_x8<<<512, 256, 0, stream>>>(W_hx, Whx_bf, (D_HID * D_IN) / 8);
  cvt_f32_to_bf16_x8<<<1024, 256, 0, stream>>>(W_hh, Whh_bf, (D_HID * D_HID) / 8);

  // 2) h_0 = 0, barrier counter = 0.
  hipMemsetAsync(h0, 0, (size_t)BATCH * D_HID * sizeof(bf16_t), stream);
  hipMemsetAsync(bar, 0, 256, stream);

  // 3) xproj = x @ W_hx^T + b_h -> bf16 [B*S, H]; pure bf16 operands.
  {
    dim3 grid(D_HID / 128, (BATCH * SEQ) / 64);
    wmma_gemm_abT<bf16_t, bf16_t, bf16_t, /*BIAS*/true, 4, 2, 4>
        <<<grid, 256, 0, stream>>>(x_bf, D_IN, Whx_bf, D_IN, xproj, D_HID, b_h, D_IN);
  }

  // 4+5) persistent recurrence (128 steps, grid barrier) + readout.
  rnn_recurrence_readout<<<128, 256, LDS_BYTES, stream>>>(
      Whh_bf, xproj, W_ph, b_p, h0, h1, out, bar);
}
